// CAM_Module_89687507075074
// MI455X (gfx1250) — matrix operand order repaired and checked by reading, not compiled, not run on hardware
//
#include <hip/hip_runtime.h>
#include <hip/hip_bf16.h>

#define B_  16
#define C_  512
#define HW_ 4096
#define C1_ 256

typedef __attribute__((ext_vector_type(16))) __bf16 v16bf;
typedef __attribute__((ext_vector_type(8)))  float  v8f;

union FragU { v16bf v; uint4 q[2]; };

// A fragment: 16x32 (MxK) bf16 tile, row-major source, ld = row stride (elems).
// Lane l (m = l&15, hi = l>>4) holds K = {hi*8..hi*8+7, 16+hi*8..16+hi*8+7} of row m.
__device__ __forceinline__ v16bf load_frag_a(const __bf16* tile, int ld, int lane) {
    int m = lane & 15, hi = lane >> 4;
    const __bf16* r = tile + (size_t)m * ld + hi * 8;
    FragU u;
    u.q[0] = *(const uint4*)(r);        // K = hi*8 .. hi*8+7
    u.q[1] = *(const uint4*)(r + 16);   // K = 16+hi*8 .. 16+hi*8+7
    return u.v;
}

// B fragment: 32x16 (KxN) bf16 tile given as B^T row-major (N x K), ld = stride
// between columns n. Lane l (n = l&15, hi = l>>4) holds K = {hi*8..hi*8+7, 16+hi*8..16+hi*8+7} of col n.
__device__ __forceinline__ v16bf load_frag_b(const __bf16* tileT, int ld, int lane) {
    int n = lane & 15, hi = lane >> 4;
    const __bf16* r = tileT + (size_t)n * ld + hi * 8;
    FragU u;
    u.q[0] = *(const uint4*)(r);        // K = hi*8 .. hi*8+7
    u.q[1] = *(const uint4*)(r + 16);   // K = 16+hi*8 .. 16+hi*8+7
    return u.v;
}

// One wave computes a 64x64 f32 tile of D = A(MxK) * B(KxN), bf16 inputs.
template<int KTOT>
__device__ __forceinline__ void wave_gemm_64x64(
    const __bf16* __restrict__ A, int lda,
    const __bf16* __restrict__ BT, int ldb,
    int m0, int n0, int lane, v8f acc[4][4]) {
    #pragma unroll 1
    for (int k = 0; k < KTOT; k += 32) {
        v16bf a[4], b[4];
        #pragma unroll
        for (int i = 0; i < 4; ++i)
            a[i] = load_frag_a(A + (size_t)(m0 + 16 * i) * lda + k, lda, lane);
        #pragma unroll
        for (int j = 0; j < 4; ++j)
            b[j] = load_frag_b(BT + (size_t)(n0 + 16 * j) * ldb + k, ldb, lane);
        if (k + 32 < KTOT) {   // wave-uniform; lowers to global_prefetch_b8
            __builtin_prefetch(A + (size_t)m0 * lda + k + 32, 0, 1);
            __builtin_prefetch(BT + (size_t)n0 * ldb + k + 32, 0, 1);
        }
        #pragma unroll
        for (int i = 0; i < 4; ++i)
            #pragma unroll
            for (int j = 0; j < 4; ++j)
                acc[i][j] = __builtin_amdgcn_wmma_f32_16x16x32_bf16(
                    false, a[i], false, b[j], (short)0, acc[i][j], false, false);
    }
}

// ---- conversion kernels -------------------------------------------------

__global__ void k_conv_w(const float* __restrict__ w, __bf16* __restrict__ wb, int n) {
    int i = blockIdx.x * blockDim.x + threadIdx.x;
    if (i < n) wb[i] = (__bf16)w[i];
}

// x (B,C,HW) f32 -> xb (B,C,HW) bf16  and  xbT (B,HW,C) bf16 (tiled LDS transpose)
__global__ void k_conv_x(const float* __restrict__ x,
                         __bf16* __restrict__ xb, __bf16* __restrict__ xbT) {
    __shared__ __bf16 t[32][33];
    int b  = blockIdx.z;
    int c0 = blockIdx.y * 32;
    int h0 = blockIdx.x * 32;
    int tx = threadIdx.x, ty = threadIdx.y;
    const float* xp = x + (size_t)b * C_ * HW_;
    __bf16* xbp = xb + (size_t)b * C_ * HW_;
    #pragma unroll
    for (int r = 0; r < 4; ++r) {
        int c = c0 + ty + 8 * r;
        float v = xp[(size_t)c * HW_ + h0 + tx];
        __bf16 bv = (__bf16)v;
        xbp[(size_t)c * HW_ + h0 + tx] = bv;
        t[ty + 8 * r][tx] = bv;
    }
    __syncthreads();
    __bf16* xtp = xbT + (size_t)b * HW_ * C_;
    #pragma unroll
    for (int r = 0; r < 4; ++r) {
        int hw = h0 + ty + 8 * r;
        xtp[(size_t)hw * C_ + c0 + tx] = t[tx][ty + 8 * r];
    }
}

// ---- GEMM 1: Q = W @ X + bias ------------------------------------------

__global__ void __launch_bounds__(256, 1)
k_gemm_q(const __bf16* __restrict__ wb, const __bf16* __restrict__ xbT,
         const float* __restrict__ bias,
         float* __restrict__ Qf, __bf16* __restrict__ Qb) {
    int wid  = (blockIdx.x * blockDim.x + threadIdx.x) >> 5;
    int lane = threadIdx.x & 31;
    int b = wid >> 8;                 // 256 tiles per batch (4 x 64)
    int r = wid & 255;
    int m0 = (r >> 6) * 64, n0 = (r & 63) * 64;
    const __bf16* A  = wb;                                // (C1, C) row-major
    const __bf16* BT = xbT + (size_t)b * HW_ * C_;        // (HW, C) row-major = X^T
    v8f acc[4][4] = {};
    wave_gemm_64x64<C_>(A, C_, BT, C_, m0, n0, lane, acc);
    int hi = lane >> 4, nc = lane & 15;
    float*  qf = Qf + (size_t)b * C1_ * HW_;
    __bf16* qb = Qb + (size_t)b * C1_ * HW_;
    #pragma unroll
    for (int i = 0; i < 4; ++i)
        #pragma unroll
        for (int j = 0; j < 4; ++j)
            #pragma unroll
            for (int v = 0; v < 8; ++v) {
                int m = m0 + 16 * i + v + 8 * hi;
                int n = n0 + 16 * j + nc;
                float q = acc[i][j][v] + bias[m];
                qf[(size_t)m * HW_ + n] = q;
                qb[(size_t)m * HW_ + n] = (__bf16)q;
            }
}

// ---- GEMM 2: E = Q @ X^T ------------------------------------------------

__global__ void __launch_bounds__(256, 1)
k_gemm_e(const __bf16* __restrict__ Qb, const __bf16* __restrict__ xb,
         float* __restrict__ E) {
    int wid  = (blockIdx.x * blockDim.x + threadIdx.x) >> 5;
    int lane = threadIdx.x & 31;
    int b = wid >> 5;                 // 32 tiles per batch (4 x 8)
    int r = wid & 31;
    int m0 = (r >> 3) * 64, n0 = (r & 7) * 64;
    const __bf16* A  = Qb + (size_t)b * C1_ * HW_;   // (C1, HW) row-major
    const __bf16* BT = xb + (size_t)b * C_ * HW_;    // B = X^T -> B^T = X (C, HW)
    v8f acc[4][4] = {};
    wave_gemm_64x64<HW_>(A, HW_, BT, HW_, m0, n0, lane, acc);
    int hi = lane >> 4, nc = lane & 15;
    float* e = E + (size_t)b * C1_ * C_;
    #pragma unroll
    for (int i = 0; i < 4; ++i)
        #pragma unroll
        for (int j = 0; j < 4; ++j)
            #pragma unroll
            for (int v = 0; v < 8; ++v) {
                int m = m0 + 16 * i + v + 8 * hi;
                int n = n0 + 16 * j + nc;
                e[(size_t)m * C_ + n] = acc[i][j][v];
            }
}

// ---- softmax over C (one wave per row of 512) ---------------------------

__global__ void k_softmax(const float* __restrict__ E, __bf16* __restrict__ attnb) {
    int row  = (blockIdx.x * blockDim.x + threadIdx.x) >> 5;
    int lane = threadIdx.x & 31;
    const float* e = E + (size_t)row * C_;
    const float s = 1.0f / 64.0f;                 // 1/sqrt(HW)
    float v[16];
    float mx = -3.4e38f;
    #pragma unroll
    for (int t = 0; t < 16; ++t) { v[t] = e[lane + 32 * t] * s; mx = fmaxf(mx, v[t]); }
    #pragma unroll
    for (int o = 16; o >= 1; o >>= 1) mx = fmaxf(mx, __shfl_xor(mx, o, 32));
    float sum = 0.0f;
    #pragma unroll
    for (int t = 0; t < 16; ++t) { v[t] = __expf(v[t] - mx); sum += v[t]; }
    #pragma unroll
    for (int o = 16; o >= 1; o >>= 1) sum += __shfl_xor(sum, o, 32);
    float inv = 1.0f / sum;
    __bf16* a = attnb + (size_t)row * C_;
    #pragma unroll
    for (int t = 0; t < 16; ++t) a[lane + 32 * t] = (__bf16)(v[t] * inv);
}

// ---- GEMM 3: out = gamma * (A @ X) + Q ----------------------------------

__global__ void __launch_bounds__(256, 1)
k_gemm_o(const __bf16* __restrict__ attnb, const __bf16* __restrict__ xbT,
         const float* __restrict__ Qf, const float* __restrict__ gamma,
         float* __restrict__ out) {
    int wid  = (blockIdx.x * blockDim.x + threadIdx.x) >> 5;
    int lane = threadIdx.x & 31;
    int b = wid >> 8;
    int r = wid & 255;
    int m0 = (r >> 6) * 64, n0 = (r & 63) * 64;
    const __bf16* A  = attnb + (size_t)b * C1_ * C_;   // (C1, C) row-major
    const __bf16* BT = xbT + (size_t)b * HW_ * C_;     // B = X -> B^T = X^T (HW, C)
    v8f acc[4][4] = {};
    wave_gemm_64x64<C_>(A, C_, BT, C_, m0, n0, lane, acc);
    float g = gamma[0];
    int hi = lane >> 4, nc = lane & 15;
    const float* qf = Qf + (size_t)b * C1_ * HW_;
    float* o = out + (size_t)b * C1_ * HW_;
    #pragma unroll
    for (int i = 0; i < 4; ++i)
        #pragma unroll
        for (int j = 0; j < 4; ++j)
            #pragma unroll
            for (int v = 0; v < 8; ++v) {
                int m = m0 + 16 * i + v + 8 * hi;
                int n = n0 + 16 * j + nc;
                o[(size_t)m * HW_ + n] = g * acc[i][j][v] + qf[(size_t)m * HW_ + n];
            }
}

// ---- launch -------------------------------------------------------------

extern "C" void kernel_launch(void* const* d_in, const int* in_sizes, int n_in,
                              void* d_out, int out_size, void* d_ws, size_t ws_size,
                              hipStream_t stream) {
    const float* x      = (const float*)d_in[0];
    const float* conv_w = (const float*)d_in[1];
    const float* conv_b = (const float*)d_in[2];
    const float* gamma  = (const float*)d_in[3];
    float* out = (float*)d_out;

    char* ws = (char*)d_ws;
    size_t off = 0;
    auto alloc = [&](size_t bytes) -> char* {
        char* p = ws + off;
        off += (bytes + 255) & ~(size_t)255;
        return p;
    };
    __bf16* wb  = (__bf16*)alloc((size_t)C1_ * C_ * 2);        // 256 KiB
    __bf16* xb  = (__bf16*)alloc((size_t)B_ * C_ * HW_ * 2);   // 64 MiB
    __bf16* xbT = (__bf16*)alloc((size_t)B_ * C_ * HW_ * 2);   // 64 MiB
    float*  Qf  = (float*) alloc((size_t)B_ * C1_ * HW_ * 4);  // 64 MiB
    __bf16* Qb  = (__bf16*)alloc((size_t)B_ * C1_ * HW_ * 2);  // 32 MiB
    float*  E   = (float*) alloc((size_t)B_ * C1_ * C_ * 4);   // 8 MiB
    __bf16* Ab  = (__bf16*)alloc((size_t)B_ * C1_ * C_ * 2);   // 4 MiB

    k_conv_w<<<(C1_ * C_ + 255) / 256, 256, 0, stream>>>(conv_w, wb, C1_ * C_);
    k_conv_x<<<dim3(HW_ / 32, C_ / 32, B_), dim3(32, 8), 0, stream>>>(x, xb, xbT);
    // 4096 waves / 8 waves per 256-thread block
    k_gemm_q<<<512, 256, 0, stream>>>(wb, xbT, conv_b, Qf, Qb);
    // 512 waves
    k_gemm_e<<<64, 256, 0, stream>>>(Qb, xb, E);
    // 4096 rows, one wave each
    k_softmax<<<512, 256, 0, stream>>>(E, Ab);
    // 4096 waves
    k_gemm_o<<<512, 256, 0, stream>>>(Ab, xbT, Qf, gamma, out);
}
